// PointEdgeSegNet_17875653886625
// MI455X (gfx1250) — compile-verified, matrix operand order checked by reading, not run on hardware
//
#include <hip/hip_runtime.h>

// =====================================================================
// PointEdgeSegNet for MI455X (gfx1250, wave32, WMMA).
// All linear layers run through a generic V_WMMA_F32_16X16X32_F16 GEMM.
// Each wave computes a 64x16 output block (4 M-tiles): weight fragments
// are loaded once per K-step and 4 independent WMMA accumulator chains
// keep the XDL pipe busy. fp32 accumulation + fp32 BN statistics.
// =====================================================================

#define BN_EPS 1e-5f
#define GEMM_MB 4   // 16-row tiles per wave (64 rows); all M here are %64==0

typedef _Float16 v16h __attribute__((ext_vector_type(16)));
typedef _Float16 v8h  __attribute__((ext_vector_type(8)));
typedef float    v8f  __attribute__((ext_vector_type(8)));

static inline dim3 g1d(size_t n, int bs) { return dim3((unsigned)((n + (size_t)bs - 1) / (size_t)bs)); }

// ---------------------------------------------------------------------
// knn (k=20) brute force: one thread per point, register insertion sort.
// Strict '>' shift keeps earlier (lower) index first on ties, matching
// jax.lax.top_k tie-breaking. Self-loop excluded (d[i,i]=inf).
// ---------------------------------------------------------------------
__global__ void knn20_kernel(const float* __restrict__ pos, int N, int* __restrict__ nbr)
{
    int i = blockIdx.x * blockDim.x + threadIdx.x;
    if (i >= N) return;
    float px = pos[3 * i + 0], py = pos[3 * i + 1], pz = pos[3 * i + 2];
    float bd[20]; int bi[20];
#pragma unroll
    for (int t = 0; t < 20; ++t) { bd[t] = 3.4e38f; bi[t] = 0; }
    for (int j = 0; j < N; ++j) {
        __builtin_prefetch(&pos[3 * (j + 32)], 0, 1);   // global_prefetch_b8
        if (j == i) continue;
        float dx = pos[3 * j + 0] - px, dy = pos[3 * j + 1] - py, dz = pos[3 * j + 2] - pz;
        float d = dx * dx + dy * dy + dz * dz;
        if (d < bd[19]) {
            int p = 19;
            while (p > 0 && bd[p - 1] > d) { bd[p] = bd[p - 1]; bi[p] = bi[p - 1]; --p; }
            bd[p] = d; bi[p] = j;
        }
    }
#pragma unroll
    for (int t = 0; t < 20; ++t) nbr[(size_t)i * 20 + t] = bi[t];
}

// knn (k=3) between two point sets + inverse-squared-distance weights.
__global__ void knn3_kernel(const float* __restrict__ posy, int Ny,
                            const float* __restrict__ posx, int Nx,
                            int* __restrict__ nbr3, float* __restrict__ w3)
{
    int y = blockIdx.x * blockDim.x + threadIdx.x;
    if (y >= Ny) return;
    float px = posy[3 * y + 0], py = posy[3 * y + 1], pz = posy[3 * y + 2];
    float bd[3] = {3.4e38f, 3.4e38f, 3.4e38f};
    int   bi[3] = {0, 0, 0};
    for (int j = 0; j < Nx; ++j) {
        float dx = posx[3 * j + 0] - px, dy = posx[3 * j + 1] - py, dz = posx[3 * j + 2] - pz;
        float d = dx * dx + dy * dy + dz * dz;
        if (d < bd[2]) {
            int p = 2;
            while (p > 0 && bd[p - 1] > d) { bd[p] = bd[p - 1]; bi[p] = bi[p - 1]; --p; }
            bd[p] = d; bi[p] = j;
        }
    }
#pragma unroll
    for (int t = 0; t < 3; ++t) {
        nbr3[y * 3 + t] = bi[t];
        w3[y * 3 + t] = 1.0f / fmaxf(bd[t], 1e-16f);
    }
}

// ---------------------------------------------------------------------
// Farthest point sampling: one block, sequential scan with LDS argmax
// (larger value wins, tie -> smaller index == jnp.argmax semantics).
// ---------------------------------------------------------------------
#define FPS_BS 1024
__global__ void fps_kernel(const float* __restrict__ pos, int N, int ns,
                           int* __restrict__ idx, float* __restrict__ mind)
{
    __shared__ float sv[FPS_BS];
    __shared__ int   si[FPS_BS];
    int tid = threadIdx.x;
    for (int i = tid; i < N; i += FPS_BS) mind[i] = 3.4e38f;
    if (tid == 0) idx[0] = 0;
    int last = 0;
    __syncthreads();
    for (int s = 1; s < ns; ++s) {
        float lx = pos[3 * last + 0], ly = pos[3 * last + 1], lz = pos[3 * last + 2];
        float bv = -1.0f; int bi = 0x7fffffff;
        for (int i = tid; i < N; i += FPS_BS) {
            float dx = pos[3 * i + 0] - lx, dy = pos[3 * i + 1] - ly, dz = pos[3 * i + 2] - lz;
            float d = dx * dx + dy * dy + dz * dz;
            float mv = fminf(mind[i], d);
            mind[i] = mv;
            if (mv > bv || (mv == bv && i < bi)) { bv = mv; bi = i; }
        }
        sv[tid] = bv; si[tid] = bi;
        __syncthreads();
        for (int off = FPS_BS / 2; off > 0; off >>= 1) {
            if (tid < off) {
                float ov = sv[tid + off]; int oi = si[tid + off];
                if (ov > sv[tid] || (ov == sv[tid] && oi < si[tid])) { sv[tid] = ov; si[tid] = oi; }
            }
            __syncthreads();
        }
        last = si[0];
        if (tid == 0) idx[s] = last;
        __syncthreads();
    }
}

__global__ void gather_rows_k(const float* __restrict__ src, const int* __restrict__ idx,
                              int M, int C, float* __restrict__ dst)
{
    size_t t = (size_t)blockIdx.x * blockDim.x + threadIdx.x;
    if (t >= (size_t)M * C) return;
    int i = (int)(t / C), c = (int)(t % C);
    dst[t] = src[(size_t)idx[i] * C + c];
}

// Edge features [x_i, x_j - x_i], zero-padded to Kpad, stored f16.
__global__ void build_edge_k(const float* __restrict__ x, const int* __restrict__ nbr,
                             int Nn, int C, int Kpad, _Float16* __restrict__ Ef)
{
    size_t tot = (size_t)Nn * 20 * Kpad;
    size_t t = (size_t)blockIdx.x * blockDim.x + threadIdx.x;
    if (t >= tot) return;
    size_t e = t / Kpad; int cc = (int)(t % Kpad);
    int i = (int)(e / 20);
    float v = 0.0f;
    if (cc < C) {
        v = x[(size_t)i * C + cc];
    } else if (cc < 2 * C) {
        int j = nbr[e];
        int c = cc - C;
        v = x[(size_t)j * C + c] - x[(size_t)i * C + c];
    }
    Ef[t] = (_Float16)v;
}

// Concat two f32 feature mats into one f16 mat, zero-padded to Kpad.
__global__ void concat_f16_k(const float* __restrict__ a, int Ca,
                             const float* __restrict__ b, int Cb,
                             int M, int Kpad, _Float16* __restrict__ out)
{
    size_t t = (size_t)blockIdx.x * blockDim.x + threadIdx.x;
    if (t >= (size_t)M * Kpad) return;
    size_t r = t / Kpad; int cc = (int)(t % Kpad);
    float v = 0.0f;
    if (cc < Ca)            v = a[r * (size_t)Ca + cc];
    else if (cc < Ca + Cb)  v = b[r * (size_t)Cb + (cc - Ca)];
    out[t] = (_Float16)v;
}

// Weight prep: W (K x N, row-major f32) -> Wt (Npad x Kpad, f16, transposed,
// zero-padded). Makes B-fragment loads contiguous b128s like A.
__global__ void prep_weight_k(const float* __restrict__ W, _Float16* __restrict__ Wt,
                              int K, int N, int Kpad, int Npad)
{
    size_t t = (size_t)blockIdx.x * blockDim.x + threadIdx.x;
    if (t >= (size_t)Npad * Kpad) return;
    int n = (int)(t / Kpad), k = (int)(t % Kpad);
    float v = (n < N && k < K) ? W[(size_t)k * N + n] : 0.0f;
    Wt[t] = (_Float16)v;
}

// ---------------------------------------------------------------------
// WMMA GEMM: C[M x Npad] = A[M x K](f16) * Wt[Npad x K]^T + bias.
// One wave per 64x16 block (GEMM_MB=4 row-tiles). Fragment layout per
// ISA 7.12.2 (16-bit A 16x32): lane L (m=L&15, h=L>>4) holds
// K = {kk+8h..kk+8h+7, kk+16+8h..+7} -> two contiguous b128 loads per
// operand per K-step. B loaded once per K-step, shared by 4 independent
// accumulator chains. C/D: VGPR r holds row (r + 8h), col m.
// ---------------------------------------------------------------------
template <bool OUTF32>
__global__ __launch_bounds__(32) void gemm16_k(const _Float16* __restrict__ A,
                                               const _Float16* __restrict__ Bt,
                                               const float* __restrict__ bias,
                                               void* __restrict__ Cout,
                                               int K, int Npad, int Nreal)
{
    int lane = threadIdx.x;
    int m = lane & 15;
    int h = lane >> 4;
    size_t tm = (size_t)blockIdx.x * (16 * GEMM_MB);
    int tn = blockIdx.y * 16;

    const _Float16* arow = A + (tm + (size_t)m) * K + 8 * h;
    const _Float16* brow = Bt + ((size_t)tn + m) * K + 8 * h;
    const size_t rstride = (size_t)16 * K;   // 16 rows of A

    v8f acc[GEMM_MB];
#pragma unroll
    for (int t = 0; t < GEMM_MB; ++t) acc[t] = (v8f){};

    for (int kk = 0; kk < K; kk += 32) {
        v8h b0 = *(const v8h*)(brow + kk);
        v8h b1 = *(const v8h*)(brow + kk + 16);
        v16h bv = __builtin_shufflevector(b0, b1, 0, 1, 2, 3, 4, 5, 6, 7,
                                          8, 9, 10, 11, 12, 13, 14, 15);
#pragma unroll
        for (int t = 0; t < GEMM_MB; ++t) {
            const _Float16* ar = arow + (size_t)t * rstride + kk;
            v8h a0 = *(const v8h*)(ar);
            v8h a1 = *(const v8h*)(ar + 16);
            v16h av = __builtin_shufflevector(a0, a1, 0, 1, 2, 3, 4, 5, 6, 7,
                                              8, 9, 10, 11, 12, 13, 14, 15);
            acc[t] = __builtin_amdgcn_wmma_f32_16x16x32_f16(false, av, false, bv,
                                                            (short)0, acc[t], false, false);
        }
    }

    int col = tn + m;
    float bb = (col < Nreal) ? bias[col] : 0.0f;
#pragma unroll
    for (int t = 0; t < GEMM_MB; ++t) {
#pragma unroll
        for (int r = 0; r < 8; ++r) {
            size_t row = tm + (size_t)t * 16 + (size_t)r + 8 * h;
            float v = acc[t][r] + bb;
            if (OUTF32) ((float*)Cout)[row * Npad + col] = v;
            else        ((_Float16*)Cout)[row * Npad + col] = (_Float16)v;
        }
    }
}

// ---------------------------------------------------------------------
// BatchNorm (training-mode batch stats, biased var) helpers.
// ---------------------------------------------------------------------
__global__ void zero_k(float* __restrict__ p, int n)
{
    int t = blockIdx.x * blockDim.x + threadIdx.x;
    if (t < n) p[t] = 0.0f;
}

__global__ void bn_stats_k(const _Float16* __restrict__ H, size_t M, int C,
                           float* __restrict__ sum, float* __restrict__ sumsq, int rpb)
{
    size_t r0 = (size_t)blockIdx.x * rpb;
    size_t r1 = r0 + rpb;
    if (r0 >= M) return;
    if (r1 > M) r1 = M;
    for (int c = threadIdx.x; c < C; c += blockDim.x) {
        float s = 0.0f, q = 0.0f;
        for (size_t r = r0; r < r1; ++r) {
            float v = (float)H[r * C + c];
            s += v; q += v * v;
        }
        atomicAdd(&sum[c], s);
        atomicAdd(&sumsq[c], q);
    }
}

__global__ void bn_relu_inplace_k(_Float16* __restrict__ H, size_t M, int C,
                                  const float* __restrict__ sum, const float* __restrict__ sumsq,
                                  const float* __restrict__ g, const float* __restrict__ be)
{
    size_t t = (size_t)blockIdx.x * blockDim.x + threadIdx.x;
    if (t >= M * (size_t)C) return;
    int c = (int)(t % C);
    float Mf = (float)M;
    float mu = sum[c] / Mf;
    float var = sumsq[c] / Mf - mu * mu;
    float rs = rsqrtf(var + BN_EPS);
    float hh = ((float)H[t] - mu) * rs * g[c] + be[c];
    H[t] = (_Float16)fmaxf(hh, 0.0f);
}

__global__ void bn_relu_f32_k(const _Float16* __restrict__ H, size_t M, int C,
                              const float* __restrict__ sum, const float* __restrict__ sumsq,
                              const float* __restrict__ g, const float* __restrict__ be,
                              float* __restrict__ out)
{
    size_t t = (size_t)blockIdx.x * blockDim.x + threadIdx.x;
    if (t >= M * (size_t)C) return;
    int c = (int)(t % C);
    float Mf = (float)M;
    float mu = sum[c] / Mf;
    float var = sumsq[c] / Mf - mu * mu;
    float rs = rsqrtf(var + BN_EPS);
    float hh = ((float)H[t] - mu) * rs * g[c] + be[c];
    out[t] = fmaxf(hh, 0.0f);
}

// Fused BN + ReLU + segmented max over the k=20 neighbor edges of a node.
__global__ void bn_relu_max_k(const _Float16* __restrict__ H, int Nn, int C,
                              const float* __restrict__ sum, const float* __restrict__ sumsq,
                              const float* __restrict__ g, const float* __restrict__ be,
                              float* __restrict__ out)
{
    size_t t = (size_t)blockIdx.x * blockDim.x + threadIdx.x;
    if (t >= (size_t)Nn * C) return;
    int i = (int)(t / C), c = (int)(t % C);
    float Mf = (float)Nn * 20.0f;
    float mu = sum[c] / Mf;
    float var = sumsq[c] / Mf - mu * mu;
    float rs = rsqrtf(var + BN_EPS);
    float gc = g[c], bc = be[c];
    float best = -3.4e38f;
#pragma unroll 4
    for (int j = 0; j < 20; ++j) {
        float hh = ((float)H[((size_t)i * 20 + j) * C + c] - mu) * rs * gc + bc;
        best = fmaxf(best, fmaxf(hh, 0.0f));
    }
    out[t] = best;
}

// knn-3 inverse-distance-weighted interpolation apply.
__global__ void interp_apply_k(const float* __restrict__ xsrc, const int* __restrict__ nbr3,
                               const float* __restrict__ w3, int Ny, int C,
                               float* __restrict__ out)
{
    size_t t = (size_t)blockIdx.x * blockDim.x + threadIdx.x;
    if (t >= (size_t)Ny * C) return;
    int y = (int)(t / C), c = (int)(t % C);
    float ws = 0.0f, acc = 0.0f;
#pragma unroll
    for (int j = 0; j < 3; ++j) {
        float w = w3[y * 3 + j];
        acc += w * xsrc[(size_t)nbr3[y * 3 + j] * C + c];
        ws += w;
    }
    out[t] = acc / ws;
}

__global__ void log_softmax_k(const float* __restrict__ lg, float* __restrict__ out,
                              int M, int NC, int ld)
{
    int r = blockIdx.x * blockDim.x + threadIdx.x;
    if (r >= M) return;
    float mx = -3.4e38f;
    for (int c = 0; c < NC; ++c) mx = fmaxf(mx, lg[(size_t)r * ld + c]);
    float s = 0.0f;
    for (int c = 0; c < NC; ++c) s += expf(lg[(size_t)r * ld + c] - mx);
    float l = logf(s);
    for (int c = 0; c < NC; ++c) out[(size_t)r * NC + c] = lg[(size_t)r * ld + c] - mx - l;
}

// =====================================================================
// Host-side orchestration helpers
// =====================================================================
struct ConvP { const float *W1, *b1, *g1, *be1, *W2, *b2, *g2, *be2; };

static void run_bn_stats(hipStream_t st, const _Float16* H, size_t M, int C,
                         float* sums, float* sumsq)
{
    zero_k<<<4, 256, 0, st>>>(sums, 1024);  // sums[512] + sumsq[512] contiguous
    const int grid = 256;
    int rpb = (int)((M + grid - 1) / grid);
    bn_stats_k<<<grid, 256, 0, st>>>(H, M, C, sums, sumsq, rpb);
}

static void run_linear_f16(hipStream_t st, const _Float16* A, size_t M, int Kpad,
                           const float* W, int Kreal, int Cout, const float* bias,
                           _Float16* Wt, _Float16* Hout)
{
    prep_weight_k<<<g1d((size_t)Cout * Kpad, 256), 256, 0, st>>>(W, Wt, Kreal, Cout, Kpad, Cout);
    gemm16_k<false><<<dim3((unsigned)(M / (16 * GEMM_MB)), (unsigned)(Cout / 16)), 32, 0, st>>>(
        A, Wt, bias, (void*)Hout, Kpad, Cout, Cout);
}

static void edgeconv(hipStream_t st, const float* xin, int C, const float* pos, int Nn,
                     const ConvP& P, int Cout, float* xout,
                     int* nbr, _Float16* Ef, _Float16* H1, _Float16* H2,
                     _Float16* Wt, float* sums, float* sumsq)
{
    knn20_kernel<<<g1d((size_t)Nn, 64), 64, 0, st>>>(pos, Nn, nbr);
    int Kpad = ((2 * C + 31) / 32) * 32;
    size_t M = (size_t)Nn * 20;
    build_edge_k<<<g1d(M * Kpad, 256), 256, 0, st>>>(xin, nbr, Nn, C, Kpad, Ef);
    // layer 1
    run_linear_f16(st, Ef, M, Kpad, P.W1, 2 * C, Cout, P.b1, Wt, H1);
    run_bn_stats(st, H1, M, Cout, sums, sumsq);
    bn_relu_inplace_k<<<g1d(M * Cout, 256), 256, 0, st>>>(H1, M, Cout, sums, sumsq, P.g1, P.be1);
    // layer 2
    run_linear_f16(st, H1, M, Cout, P.W2, Cout, Cout, P.b2, Wt, H2);
    run_bn_stats(st, H2, M, Cout, sums, sumsq);
    bn_relu_max_k<<<g1d((size_t)Nn * Cout, 256), 256, 0, st>>>(H2, Nn, Cout, sums, sumsq,
                                                               P.g2, P.be2, xout);
}

static void interp(hipStream_t st, const float* xsrc, const float* posx, int Nx,
                   const float* posy, int Ny, int C, float* outy, int* nbr3, float* w3)
{
    knn3_kernel<<<g1d((size_t)Ny, 64), 64, 0, st>>>(posy, Ny, posx, Nx, nbr3, w3);
    interp_apply_k<<<g1d((size_t)Ny * C, 256), 256, 0, st>>>(xsrc, nbr3, w3, Ny, C, outy);
}

static void dec_layer(hipStream_t st, const float* a, int Ca, const float* b, int Cb,
                      int M, const float* W, const float* bias, const float* g, const float* be,
                      int Cout, float* out, _Float16* Ef, _Float16* H, _Float16* Wt,
                      float* sums, float* sumsq)
{
    int Kreal = Ca + Cb;
    int Kpad = ((Kreal + 31) / 32) * 32;
    concat_f16_k<<<g1d((size_t)M * Kpad, 256), 256, 0, st>>>(a, Ca, b, Cb, M, Kpad, Ef);
    run_linear_f16(st, Ef, (size_t)M, Kpad, W, Kreal, Cout, bias, Wt, H);
    run_bn_stats(st, H, (size_t)M, Cout, sums, sumsq);
    bn_relu_f32_k<<<g1d((size_t)M * Cout, 256), 256, 0, st>>>(H, (size_t)M, Cout,
                                                              sums, sumsq, g, be, out);
}

// =====================================================================
extern "C" void kernel_launch(void* const* d_in, const int* in_sizes, int n_in,
                              void* d_out, int out_size, void* d_ws, size_t ws_size,
                              hipStream_t stream)
{
    (void)in_sizes; (void)n_in; (void)out_size; (void)ws_size;

    const int N0 = 8192, N1 = 2048, N2 = 512, N3 = 128;

    // ---- inputs (setup_inputs() dict order, params flattened in insertion order) ----
    const float* x_in   = (const float*)d_in[0];   // [8192 x 6]
    const float* pos_in = (const float*)d_in[1];   // [8192 x 3]
    // d_in[2] = batch (single graph, unused)
    int q = 3;
    auto F = [&](void) { return (const float*)d_in[q++]; };
    ConvP cp1; cp1.W1=F(); cp1.b1=F(); cp1.g1=F(); cp1.be1=F(); cp1.W2=F(); cp1.b2=F(); cp1.g2=F(); cp1.be2=F();
    ConvP cp2; cp2.W1=F(); cp2.b1=F(); cp2.g1=F(); cp2.be1=F(); cp2.W2=F(); cp2.b2=F(); cp2.g2=F(); cp2.be2=F();
    ConvP cp3; cp3.W1=F(); cp3.b1=F(); cp3.g1=F(); cp3.be1=F(); cp3.W2=F(); cp3.b2=F(); cp3.g2=F(); cp3.be2=F();
    ConvP cp4; cp4.W1=F(); cp4.b1=F(); cp4.g1=F(); cp4.be1=F(); cp4.W2=F(); cp4.b2=F(); cp4.g2=F(); cp4.be2=F();
    const float *d1W=F(), *d1b=F(), *d1g=F(), *d1be=F();
    const float *d2W=F(), *d2b=F(), *d2g=F(), *d2be=F();
    const float *d3W=F(), *d3b=F(), *d3g=F(), *d3be=F();
    const float *hW1=F(), *hb1=F(), *hg1=F(), *hbe1=F(), *hW2=F(), *hb2=F();

    // ---- workspace carve-out (256B aligned regions) ----
    char* wsb = (char*)d_ws;
    size_t off = 0;
    auto A = [&](size_t bytes) -> char* {
        char* p = wsb + off;
        off += (bytes + 255) & ~(size_t)255;
        return p;
    };
    float* x1   = (float*)A((size_t)N0 * 64 * 4);
    float* x2   = (float*)A((size_t)N1 * 128 * 4);
    float* x3   = (float*)A((size_t)N2 * 256 * 4);
    float* x4   = (float*)A((size_t)N3 * 512 * 4);
    float* pos1 = (float*)A((size_t)N1 * 3 * 4);
    float* pos2 = (float*)A((size_t)N2 * 3 * 4);
    float* pos3 = (float*)A((size_t)N3 * 3 * 4);
    int*   idx1 = (int*)A((size_t)N1 * 4);
    int*   idx2 = (int*)A((size_t)N2 * 4);
    int*   idx3 = (int*)A((size_t)N3 * 4);
    float* xs   = (float*)A((size_t)N1 * 64 * 4);       // subsampled features (max 2048x64)
    float* mind = (float*)A((size_t)N0 * 4);
    int*   nbr  = (int*)A((size_t)N0 * 20 * 4);
    int*   nbr3 = (int*)A((size_t)N0 * 3 * 4);
    float* w3   = (float*)A((size_t)N0 * 3 * 4);
    float* up2  = (float*)A((size_t)N2 * 512 * 4);
    float* up1  = (float*)A((size_t)N1 * 256 * 4);
    float* up0  = (float*)A((size_t)N0 * 128 * 4);
    float* d2f  = (float*)A((size_t)N2 * 256 * 4);
    float* d1f  = (float*)A((size_t)N1 * 128 * 4);
    float* d0f  = (float*)A((size_t)N0 * 64 * 4);
    float* hf   = (float*)A((size_t)N0 * 64 * 4);
    float* sums = (float*)A(1024 * 4);
    float* sumsq = sums + 512;
    _Float16* Wt = (_Float16*)A((size_t)512 * 1024 * 2);    // max transposed weight
    _Float16* Ef = (_Float16*)A((size_t)N0 * 20 * 32 * 2);  // 10.5 MB, covers all A-mats
    _Float16* H1 = (_Float16*)A((size_t)N0 * 20 * 64 * 2);  // 21 MB
    _Float16* H2 = (_Float16*)A((size_t)N0 * 20 * 64 * 2);  // 21 MB
    float* logits = (float*)A((size_t)N0 * 16 * 4);

    // ================= encoder =================
    // conv1 on (x, pos), C=6 -> 64
    edgeconv(stream, x_in, 6, pos_in, N0, cp1, 64, x1, nbr, Ef, H1, H2, Wt, sums, sumsq);

    // FPS 8192 -> 2048; gather pos1, x1s
    fps_kernel<<<1, FPS_BS, 0, stream>>>(pos_in, N0, N1, idx1, mind);
    gather_rows_k<<<g1d((size_t)N1 * 3, 256), 256, 0, stream>>>(pos_in, idx1, N1, 3, pos1);
    gather_rows_k<<<g1d((size_t)N1 * 64, 256), 256, 0, stream>>>(x1, idx1, N1, 64, xs);
    // conv2: 64 -> 128
    edgeconv(stream, xs, 64, pos1, N1, cp2, 128, x2, nbr, Ef, H1, H2, Wt, sums, sumsq);

    // FPS 2048 -> 512
    fps_kernel<<<1, FPS_BS, 0, stream>>>(pos1, N1, N2, idx2, mind);
    gather_rows_k<<<g1d((size_t)N2 * 3, 256), 256, 0, stream>>>(pos1, idx2, N2, 3, pos2);
    gather_rows_k<<<g1d((size_t)N2 * 128, 256), 256, 0, stream>>>(x2, idx2, N2, 128, xs);
    // conv3: 128 -> 256
    edgeconv(stream, xs, 128, pos2, N2, cp3, 256, x3, nbr, Ef, H1, H2, Wt, sums, sumsq);

    // FPS 512 -> 128
    fps_kernel<<<1, FPS_BS, 0, stream>>>(pos2, N2, N3, idx3, mind);
    gather_rows_k<<<g1d((size_t)N3 * 3, 256), 256, 0, stream>>>(pos2, idx3, N3, 3, pos3);
    gather_rows_k<<<g1d((size_t)N3 * 256, 256), 256, 0, stream>>>(x3, idx3, N3, 256, xs);
    // conv4: 256 -> 512
    edgeconv(stream, xs, 256, pos3, N3, cp4, 512, x4, nbr, Ef, H1, H2, Wt, sums, sumsq);

    // ================= decoder =================
    // up2 = knn_interp(x4: pos3 -> pos2); d2 = lin_bn_relu([up2, x3]) : 768 -> 256
    interp(stream, x4, pos3, N3, pos2, N2, 512, up2, nbr3, w3);
    dec_layer(stream, up2, 512, x3, 256, N2, d1W, d1b, d1g, d1be, 256, d2f,
              Ef, H1, Wt, sums, sumsq);

    // up1 = knn_interp(d2: pos2 -> pos1); d1 = lin_bn_relu([up1, x2]) : 384 -> 128
    interp(stream, d2f, pos2, N2, pos1, N1, 256, up1, nbr3, w3);
    dec_layer(stream, up1, 256, x2, 128, N1, d2W, d2b, d2g, d2be, 128, d1f,
              Ef, H1, Wt, sums, sumsq);

    // up0 = knn_interp(d1: pos1 -> pos); d0 = lin_bn_relu([up0, x1]) : 192 -> 64
    interp(stream, d1f, pos1, N1, pos_in, N0, 128, up0, nbr3, w3);
    dec_layer(stream, up0, 128, x1, 64, N0, d3W, d3b, d3g, d3be, 64, d0f,
              Ef, H1, Wt, sums, sumsq);

    // head layer 1: [d0, x] (K=70 -> pad 96) -> 64, BN+ReLU
    dec_layer(stream, d0f, 64, x_in, 6, N0, hW1, hb1, hg1, hbe1, 64, hf,
              Ef, H1, Wt, sums, sumsq);

    // head layer 2: logits = h @ W2 + b2, (64 -> 13, Npad=16), fp32 out
    concat_f16_k<<<g1d((size_t)N0 * 64, 256), 256, 0, stream>>>(hf, 64, (const float*)0, 0,
                                                                N0, 64, Ef);
    prep_weight_k<<<g1d((size_t)16 * 64, 256), 256, 0, stream>>>(hW2, Wt, 64, 13, 64, 16);
    gemm16_k<true><<<dim3(N0 / (16 * GEMM_MB), 1), 32, 0, stream>>>(Ef, Wt, hb2, (void*)logits,
                                                                    64, 16, 13);

    // log_softmax over 13 classes -> d_out [8192 x 13] f32
    log_softmax_k<<<g1d((size_t)N0, 256), 256, 0, stream>>>(logits, (float*)d_out, N0, 13, 16);
}
